// TAMKOT_31954556682911
// MI455X (gfx1250) — matrix operand order repaired and checked by reading, not compiled, not run on hardware
//
#include <hip/hip_runtime.h>
#include <hip/hip_bf16.h>

typedef __bf16 v16bf __attribute__((ext_vector_type(16)));
typedef float  v8f   __attribute__((ext_vector_type(8)));

#define BB 256     // batch
#define SS 500     // sequence length
#define TT 498     // output steps (S-2)
#define HH 128     // hidden
#define KA 704     // GEMM K: 512 (h4) + 128 (qa) + 64 (l_e)
#define KTN 22     // KA / 32
#define NNW 512    // GEMM N: 4 gates x 128
#define BT 16      // batch tile per workgroup

// ---------------------------------------------------------------------------
// Kernel 1: pack Wcat(704 x 512) bf16 into WMMA B-fragment order.
//   Packed index = ((ntile*KTN + ktile)*32 + lane)*16 + e
//   B layout: lane<16 -> N=lane, Klocal=e ; lane>=16 -> N=lane-16, Klocal=e+16
//   K<512: Wh[g, ks, j, i] (ks=K/128, j=K%128); K<640: WQ[g, K-512, i];
//   else: WL[g, K-640, i]; with g=N/128, i=N%128.
// ---------------------------------------------------------------------------
__global__ __launch_bounds__(256) void tamkot_pack_kernel(
    const float* __restrict__ WQ, const float* __restrict__ WL,
    const float* __restrict__ Wh, unsigned short* __restrict__ Wcat_raw)
{
    __bf16* Wcat = (__bf16*)Wcat_raw;
    int idx = blockIdx.x * 256 + threadIdx.x;
    const int total = (NNW / 16) * KTN * 32 * 16;   // 360448
    if (idx >= total) return;
    int e    = idx & 15;
    int lane = (idx >> 4) & 31;
    int kt   = (idx >> 9) % KTN;
    int nt   = (idx >> 9) / KTN;
    int K = kt * 32 + e + ((lane >= 16) ? 16 : 0);
    int N = nt * 16 + (lane & 15);
    int g = N >> 7, i = N & 127;
    float v;
    if (K < 512) {
        int ks = K >> 7, j = K & 127;
        v = Wh[(((g * 4 + ks) * 128) + j) * 128 + i];
    } else if (K < 640) {
        v = WQ[(g * 128 + (K - 512)) * 128 + i];
    } else {
        v = WL[(g * 64 + (K - 640)) * 128 + i];
    }
    Wcat[idx] = (__bf16)v;
}

__device__ __forceinline__ float sigf(float x) { return 1.0f / (1.0f + __expf(-x)); }

// ---------------------------------------------------------------------------
// Kernel 2: persistent recurrent kernel. 16 blocks x 256 threads (8 wave32s).
// Each block owns a 16-row batch tile and runs all 498 steps locally.
// Per step: build A(16x704) bf16 in LDS (A-fragment packed order), then
// pre(16x512) = A @ Wcat via v_wmma_f32_16x16x32_bf16, then gates + preds.
// B streams from L2 every step (721 KB, permanently L2-hot). The base pointer
// is laundered through asm each iteration (prevents hoist+spill of the
// loop-invariant fragments) and cast to address_space(1) so the loads lower
// to global_load_b128 (GVS: sgpr base + lane offset + imm), not flat_load.
// ---------------------------------------------------------------------------
__global__ __launch_bounds__(256) void tamkot_recurrent_kernel(
    const int*   __restrict__ q_data, const int* __restrict__ a_data,
    const int*   __restrict__ l_data, const float* __restrict__ d_data,
    const float* __restrict__ q_embed, const float* __restrict__ a_embed,
    const float* __restrict__ l_embed,
    const float* __restrict__ bQ, const float* __restrict__ bL,
    const float* __restrict__ W_Qh, const float* __restrict__ b_Qh,
    const float* __restrict__ W_Lh, const float* __restrict__ b_Lh,
    const unsigned short* __restrict__ Wcat_raw, float* __restrict__ out)
{
    __shared__ __bf16 Apack[KTN * 32 * 16];   // 22.0 KB: A-fragments, packed
    __shared__ float  preS[BT * NNW];         // 32 KB: GEMM result
    __shared__ float  hS[BT * HH];            // 8 KB
    __shared__ float  mS[BT * HH];            // 8 KB
    __shared__ float  dS[BT], omdS[BT];
    __shared__ int    kselS[BT], qS[BT], aS[BT], lS[BT];

    const int tid  = threadIdx.x;
    const int b0   = blockIdx.x * BT;
    const int lane = tid & 31;
    const int w    = tid >> 5;                // wave id 0..7

    for (int p = tid; p < BT * HH; p += 256) { hS[p] = 0.f; mS[p] = 0.f; }
    __syncthreads();

    for (int s = 1; s <= TT; ++s) {
        // ---- phase 0: per-row scalars (d, d1 -> k-selector; gather indices)
        if (tid < BT) {
            int b = b0 + tid;
            float d  = d_data[b * SS + s];
            float d1 = d_data[b * SS + s - 1];
            dS[tid] = d; omdS[tid] = 1.f - d;
            int di = (d != 0.f), d1i = (d1 != 0.f);
            // c = [(1-d)(1-d1), d*d1, d(1-d1), d1(1-d)] -> exactly one k hits
            kselS[tid] = di ? (d1i ? 1 : 2) : (d1i ? 3 : 0);
            qS[tid] = q_data[b * SS + s];
            aS[tid] = a_data[b * SS + s];
            lS[tid] = l_data[b * SS + s];
        }
        __syncthreads();

        // ---- phase 1: build packed A (16x704) = [h4 | (1-d)*qa | d*l_e]
        // A-fragment layout: lane<16 -> M=lane, Klocal = e
        //                    lane>=16 -> M=lane-16, Klocal = e+16
        #pragma unroll
        for (int ii = 0; ii < (KTN * 32 * 16) / 256; ++ii) {   // 44 per thread
            int idx = ii * 256 + tid;
            int e  = idx & 15;
            int ln = (idx >> 4) & 31;
            int kt = idx >> 9;
            int M  = ln & 15;
            int K  = kt * 32 + e + ((ln >= 16) ? 16 : 0);
            float v;
            if (K < 512) {
                int ks = K >> 7, j = K & 127;
                v = (ks == kselS[M]) ? hS[M * HH + j] : 0.f;
            } else if (K < 576) {
                v = omdS[M] * q_embed[qS[M] * 64 + (K - 512)];
            } else if (K < 640) {
                v = omdS[M] * a_embed[aS[M] * 64 + (K - 576)];
            } else {
                v = dS[M] * l_embed[lS[M] * 64 + (K - 640)];
            }
            Apack[idx] = (__bf16)v;
        }
        __syncthreads();

        // ---- phase 2: pre(16x512) = A(16x704) @ Wcat(704x512), WMMA bf16
        // Launder the B base pointer so loads stay inside the s-loop (served
        // from L2 each step, no hoist->spill), and force address_space(1) so
        // they lower to global_load_b128 instead of flat_load_b128.
        unsigned long long wp = (unsigned long long)Wcat_raw;
        asm volatile("" : "+s"(wp));
        typedef __attribute__((address_space(1))) const v16bf ASbf;
        ASbf* Wv = (ASbf*)wp + (size_t)(w * 4) * (KTN * 32) + lane;

        v8f acc[4];
        #pragma unroll
        for (int nt = 0; nt < 4; ++nt)
            #pragma unroll
            for (int r = 0; r < 8; ++r) acc[nt][r] = 0.f;

        #pragma unroll 2
        for (int kt = 0; kt < KTN; ++kt) {
            v16bf aF = *(const v16bf*)&Apack[(kt * 32 + lane) * 16];
            #pragma unroll
            for (int nt = 0; nt < 4; ++nt) {
                v16bf bF = Wv[(nt * KTN + kt) * 32];
                acc[nt] = __builtin_amdgcn_wmma_f32_16x16x32_bf16(
                    false, aF, false, bF, (short)0, acc[nt], false, false);
            }
        }
        // C/D layout: VGPR r, lane<16 -> M=r ; lane>=16 -> M=r+8 ; N = n0+lane%16
        #pragma unroll
        for (int nt = 0; nt < 4; ++nt) {
            #pragma unroll
            for (int r = 0; r < 8; ++r) {
                int M = r + ((lane >= 16) ? 8 : 0);
                int N = (w * 4 + nt) * 16 + (lane & 15);
                preS[M * NNW + N] = acc[nt][r];
            }
        }
        __syncthreads();

        // ---- phase 3: blended bias + gates + state update (f32)
        #pragma unroll
        for (int ii = 0; ii < 8; ++ii) {
            int p = tid * 8 + ii;             // p = M*128 + j
            int M = p >> 7, j = p & 127;
            float d = dS[M], omd = omdS[M];
            float p0 = preS[M * NNW +   0 + j] + omd * bQ[      j] + d * bL[      j];
            float p1 = preS[M * NNW + 128 + j] + omd * bQ[128 + j] + d * bL[128 + j];
            float p2 = preS[M * NNW + 256 + j] + omd * bQ[256 + j] + d * bL[256 + j];
            float p3 = preS[M * NNW + 384 + j] + omd * bQ[384 + j] + d * bL[384 + j];
            float ig = sigf(p0);
            float gg = tanhf(p1);
            float fg = sigf(p2);
            float og = sigf(p3);
            float mn = fg * mS[p] + ig * gg;
            float hn = og * tanhf(mn);
            mS[p] = mn;
            hS[p] = hn;
        }
        __syncthreads();

        // ---- phase 4: predictions (tiny dot products, one row per thread)
        if (tid < BT) {
            int M = tid, b = b0 + M;
            float s1 = b_Lh[0], s2 = b_Qh[0];
            for (int j = 0; j < HH; ++j) {
                float h = hS[M * HH + j];
                s1 += h * W_Lh[j];
                s2 += h * W_Qh[j];
            }
            int qn = q_data[b * SS + s + 1];
            for (int e2 = 0; e2 < 64; ++e2)
                s2 += q_embed[qn * 64 + e2] * W_Qh[HH + e2];
            int o = s - 1;
            out[b * TT + o]           = sigf(s2);   // pred_q
            out[BB * TT + b * TT + o] = sigf(s1);   // pred_type
        }
        __syncthreads();
    }
}

extern "C" void kernel_launch(void* const* d_in, const int* in_sizes, int n_in,
                              void* d_out, int out_size, void* d_ws, size_t ws_size,
                              hipStream_t stream) {
    const int*   q_data  = (const int*)  d_in[0];
    const int*   a_data  = (const int*)  d_in[1];
    const int*   l_data  = (const int*)  d_in[2];
    const float* d_data  = (const float*)d_in[3];
    const float* q_embed = (const float*)d_in[4];
    const float* a_embed = (const float*)d_in[5];
    const float* l_embed = (const float*)d_in[6];
    const float* WQ      = (const float*)d_in[7];
    const float* bQ      = (const float*)d_in[8];
    const float* WL      = (const float*)d_in[9];
    const float* bL      = (const float*)d_in[10];
    const float* Wh      = (const float*)d_in[11];
    const float* W_Qh    = (const float*)d_in[12];
    const float* b_Qh    = (const float*)d_in[13];
    const float* W_Lh    = (const float*)d_in[14];
    const float* b_Lh    = (const float*)d_in[15];
    float* out = (float*)d_out;
    unsigned short* Wcat = (unsigned short*)d_ws;   // 704*512 bf16 = 721 KB

    const int packTotal = (NNW / 16) * KTN * 32 * 16;   // 360448
    tamkot_pack_kernel<<<(packTotal + 255) / 256, 256, 0, stream>>>(WQ, WL, Wh, Wcat);
    tamkot_recurrent_kernel<<<BB / BT, 256, 0, stream>>>(
        q_data, a_data, l_data, d_data, q_embed, a_embed, l_embed,
        bQ, bL, W_Qh, b_Qh, W_Lh, b_Lh, Wcat, out);
}
